// GCNModel_50491635532468
// MI455X (gfx1250) — matrix operand order repaired and checked by reading, not compiled, not run on hardware
//
#include <hip/hip_runtime.h>
#include <hip/hip_bf16.h>

// ---------------------------------------------------------------------------
// GraphMAE GCN forward on MI455X (gfx1250, wave32).
//  - Dense GEMMs: v_wmma_f32_16x16x32_bf16, operands pre-converted/pre-swizzled
//    so the inner loop is pure b128 loads + WMMA; branch-free epilogue with
//    immediate-offset stores.
//  - Edge aggregation: global_atomic_add_f32 scatter (memory-bound phase,
//    ~3.7GB traffic -> ~160us at 23.3TB/s; working set is L2-resident).
// ---------------------------------------------------------------------------

#define NN     50000
#define NE     800000
#define DIM    128
#define NLAY   3
#define NMASK  15000
#define FEATS  (NN * DIM)       // 6,400,000
#define WMAT   (DIM * DIM)      // 16,384
#define NMAT   4                // 3 layer weights + decoder

typedef __attribute__((ext_vector_type(16))) __bf16 v16bf;
typedef __attribute__((ext_vector_type(8)))  __bf16 v8bf;
typedef __attribute__((ext_vector_type(8)))  float  v8f;

__device__ __forceinline__ __bf16 f2bf(float f) { return (__bf16)f; }

// ---------------- generic helpers ----------------

__global__ void zero_f32(float* __restrict__ p, int n) {
    int i = blockIdx.x * blockDim.x + threadIdx.x;
    if (i < n) p[i] = 0.0f;
}

// ---------------- degrees / norms ----------------

__global__ void deg_kernel(const int* __restrict__ src, const int* __restrict__ dst,
                           float* __restrict__ odeg, float* __restrict__ ideg, int E) {
    int e = blockIdx.x * blockDim.x + threadIdx.x;
    if (e < E) {
        atomicAdd(&odeg[src[e]], 1.0f);
        atomicAdd(&ideg[dst[e]], 1.0f);
    }
}

__global__ void norm_kernel(float* __restrict__ ns, float* __restrict__ nd, int N) {
    int n = blockIdx.x * blockDim.x + threadIdx.x;
    if (n < N) {
        ns[n] = rsqrtf(fmaxf(ns[n], 1.0f));
        nd[n] = rsqrtf(fmaxf(nd[n], 1.0f));
    }
}

// ---------------- h0 = bf16(attr), then masked rows = bf16(token) ----------

__global__ void init_h_bf16(const float* __restrict__ attr, __bf16* __restrict__ hbf, int n) {
    int i = blockIdx.x * blockDim.x + threadIdx.x;
    if (i < n) hbf[i] = f2bf(attr[i]);
}

__global__ __launch_bounds__(DIM)
void mask_kernel(__bf16* __restrict__ hbf, const int* __restrict__ mn,
                 const float* __restrict__ tok, int NM) {
    int b = blockIdx.x;
    if (b < NM) hbf[(size_t)mn[b] * DIM + threadIdx.x] = f2bf(tok[threadIdx.x]);
}

// ---------------- weight pre-swizzle -------------------------------------
// Wsw[mat][kb4][tn][lane][idx] = bf16( W[mat][kb4*32 + (idx&7) + 8*(lane>>4) + 16*(idx>>3)]
//                                            [tn*16 + (lane&15)] )
// -> each lane's B fragment (v16bf, K = (idx&7) + 8*half + 16*(idx>>3)) is one contiguous
//    32-byte chunk.

__global__ void prep_w(const float* __restrict__ Ws, const float* __restrict__ decW,
                       __bf16* __restrict__ wsw) {
    int i = blockIdx.x * blockDim.x + threadIdx.x;
    if (i >= NMAT * WMAT) return;
    const int mat = i >> 14;
    const int r   = i & (WMAT - 1);
    const int idx = r & 15;
    const int lane = (r >> 4) & 31;
    const int tn   = (r >> 9) & 7;
    const int kb4  = r >> 12;
    const int k = kb4 * 32 + (idx & 7) + 8 * (lane >> 4) + 16 * (idx >> 3);
    const int n = tn * 16 + (lane & 15);
    const float* W = (mat < NLAY) ? (Ws + mat * WMAT) : decW;
    wsw[i] = f2bf(W[k * DIM + n]);
}

// ---------------- WMMA GEMM: C[M,128] = A[M,128] * B[128,128] + bias -------
// Block = 256 threads = 8 waves; block owns a 16-row stripe, wave w owns the
// 16-col tile at 16*w.  A row-major bf16: lane (half,m) fragment = two
// contiguous 8-elem segments (ISA 7.12.2 16-bit A 16x32 layout).  B comes
// pre-swizzled so each lane's fragment is contiguous.  M is always a multiple
// of 16 and bias is always a valid pointer -> fully branch-free.
// C/D layout: vgpr r: lanes 0-15 -> (M=r, N=lane); lanes 16-31 -> (M=8+r).

__global__ __launch_bounds__(256)
void gemm_wmma_bf16(const __bf16* __restrict__ Abf, const __bf16* __restrict__ Bsw,
                    const float* __restrict__ bias, float* __restrict__ C) {
    const int tile_m = blockIdx.x;
    const int wave   = threadIdx.x >> 5;     // N-tile index, 0..7
    const int lane   = threadIdx.x & 31;
    const int half   = lane >> 4;
    const int lm     = lane & 15;
    const int row0   = tile_m * 16;
    const __bf16* arow = Abf + (size_t)(row0 + lm) * DIM;

    v8f acc = {};
    #pragma unroll
    for (int kb4 = 0; kb4 < 4; ++kb4) {
        const int kb = kb4 * 32;
        const v8bf alo = *(const v8bf*)(arow + kb + 8 * half);
        const v8bf ahi = *(const v8bf*)(arow + kb + 16 + 8 * half);
        const v16bf a = __builtin_shufflevector(alo, ahi,
                            0,1,2,3,4,5,6,7,8,9,10,11,12,13,14,15);
        const v16bf b = *(const v16bf*)(Bsw + (((kb4 * 8 + wave) * 32 + lane) << 4));
        acc = __builtin_amdgcn_wmma_f32_16x16x32_bf16(
                  false, a, false, b, (short)0, acc, false, false);
    }

    const int col = wave * 16 + lm;
    const float bv = bias[col];
    float* cptr = C + (size_t)(row0 + 8 * half) * DIM + col;
    #pragma unroll
    for (int r = 0; r < 8; ++r) {
        cptr[r * DIM] = acc[r] + bv;     // 8 stores, constant immediate offsets
    }
}

// ---------------- edge scatter-add ----------------
// 2 edges per 256-thread block; both edges' indices are block-uniform
// (scalarizable loads), 128 consecutive lanes stream one edge's feature row
// (coalesced 512B), scatter via global_atomic_add_f32.  NE is even and the
// grid covers it exactly -> no bounds checks.

__global__ __launch_bounds__(256)
void edge_kernel(const float* __restrict__ t, const float* __restrict__ ns,
                 const int* __restrict__ src, const int* __restrict__ dst,
                 float* __restrict__ agg) {
    const int e0  = blockIdx.x * 2;
    const int sub = threadIdx.x >> 7;            // 0 or 1
    const int f   = threadIdx.x & (DIM - 1);
    const int2 sp = *(const int2*)(src + e0);    // block-uniform -> s_load
    const int2 dp = *(const int2*)(dst + e0);
    const int s = sub ? sp.y : sp.x;
    const int d = sub ? dp.y : dp.x;
    atomicAdd(&agg[(size_t)d * DIM + f], t[(size_t)s * DIM + f] * ns[s]);
}

// ---------------- h = bf16(relu(agg * norm_dst + bias)) ----------------

__global__ void relu_kernel(const float* __restrict__ agg, const float* __restrict__ nd,
                            const float* __restrict__ b, __bf16* __restrict__ hbf, int total) {
    int i = blockIdx.x * blockDim.x + threadIdx.x;
    if (i < total) {
        const int n = i >> 7;
        const int f = i & (DIM - 1);
        hbf[i] = f2bf(fmaxf(agg[i] * nd[n] + b[f], 0.0f));
    }
}

// ---------------- masked MSE ----------------

__global__ __launch_bounds__(DIM)
void loss_kernel(const float* __restrict__ recon, const float* __restrict__ attr,
                 const int* __restrict__ mn, float* __restrict__ accum, int NM) {
    __shared__ float s[DIM];
    const int b = blockIdx.x;
    const int f = threadIdx.x;
    if (b >= NM) return;
    const int n = mn[b];
    const float d = recon[(size_t)n * DIM + f] - attr[(size_t)n * DIM + f];
    s[f] = d * d;
    __syncthreads();
    #pragma unroll
    for (int st = DIM / 2; st > 0; st >>= 1) {
        if (f < st) s[f] += s[f + st];
        __syncthreads();
    }
    if (f == 0) atomicAdd(accum, s[0]);
}

__global__ void fin_kernel(const float* __restrict__ accum, float* __restrict__ out) {
    out[0] = accum[0] * (1.0f / ((float)NMASK * (float)DIM));
}

// ---------------------------------------------------------------------------

extern "C" void kernel_launch(void* const* d_in, const int* in_sizes, int n_in,
                              void* d_out, int out_size, void* d_ws, size_t ws_size,
                              hipStream_t stream) {
    const float* attr  = (const float*)d_in[0];   // [NN, DIM]
    const int*   src   = (const int*)  d_in[1];   // [NE]
    const int*   dst   = (const int*)  d_in[2];   // [NE]
    const float* Ws    = (const float*)d_in[3];   // [NLAY, DIM, DIM]
    const float* bs    = (const float*)d_in[4];   // [NLAY, DIM]
    const float* decW  = (const float*)d_in[5];   // [DIM, DIM]
    const float* decb  = (const float*)d_in[6];   // [DIM]
    const float* tok   = (const float*)d_in[7];   // [1, DIM]
    const int*   mn    = (const int*)  d_in[8];   // [NMASK]
    float* out = (float*)d_out;

    // workspace: [hbf FEATS bf16][wsw 4*WMAT bf16][t][agg][ns][nd][accum][zbias]
    __bf16* hbf = (__bf16*)d_ws;                        // 12.8 MB
    __bf16* wsw = hbf + FEATS;                          // 128 KB
    float*  t   = (float*)(wsw + NMAT * WMAT);          // 25.6 MB
    float*  agg = t + FEATS;                            // 25.6 MB
    float*  ns  = agg + FEATS;                          // NN
    float*  nd  = ns + NN;                              // NN
    float*  accum = nd + NN;                            // 1
    float*  zbias = accum + 1;                          // DIM (stays zero)

    const int THR = 256;
    const int gF  = (FEATS + THR - 1) / THR;
    const int gN  = (NN + THR - 1) / THR;
    const int gE  = (NE + THR - 1) / THR;
    const int gW  = (NMAT * WMAT + THR - 1) / THR;
    const int tilesM = NN / 16;                         // 3125 (exact)
    const int nzero  = 2 * NN + 1 + DIM;                // ns|nd|accum|zbias

    // 1) degrees -> norms  (zero pass also clears accumulator + zero-bias)
    zero_f32<<<(nzero + THR - 1) / THR, THR, 0, stream>>>(ns, nzero);
    deg_kernel<<<gE, THR, 0, stream>>>(src, dst, ns, nd, NE);
    norm_kernel<<<gN, THR, 0, stream>>>(ns, nd, NN);

    // 2) weights -> swizzled bf16; h0 -> bf16 with masked rows
    prep_w<<<gW, THR, 0, stream>>>(Ws, decW, wsw);
    init_h_bf16<<<gF, THR, 0, stream>>>(attr, hbf, FEATS);
    mask_kernel<<<NMASK, DIM, 0, stream>>>(hbf, mn, tok, NMASK);

    // 3) GCN layers
    for (int l = 0; l < NLAY; ++l) {
        gemm_wmma_bf16<<<tilesM, 256, 0, stream>>>(hbf, wsw + l * WMAT, zbias, t);
        zero_f32<<<gF, THR, 0, stream>>>(agg, FEATS);
        edge_kernel<<<NE / 2, 256, 0, stream>>>(t, ns, src, dst, agg);
        relu_kernel<<<gF, THR, 0, stream>>>(agg, nd, bs + l * DIM, hbf, FEATS);
    }

    // 4) decoder + masked MSE
    gemm_wmma_bf16<<<tilesM, 256, 0, stream>>>(hbf, wsw + NLAY * WMAT, decb, t);
    loss_kernel<<<NMASK, DIM, 0, stream>>>(t, attr, mn, accum, NMASK);
    fin_kernel<<<1, 1, 0, stream>>>(accum, out);
}
